// NonAttention_EncDec_27075473834736
// MI455X (gfx1250) — matrix operand order repaired and checked by reading, not compiled, not run on hardware
//
#include <hip/hip_runtime.h>
#include <hip/hip_bf16.h>

#define SEQ_LEN 512
#define INPUT_SIZE 4096
#define HIDDEN 2048
#define GATES (3 * HIDDEN)   /* 6144 */
#define OUTPUT_SIZE 4096
#define STEPS 30

typedef __bf16 bf16_t;
typedef __attribute__((ext_vector_type(16))) __bf16 v16bf;
typedef __attribute__((ext_vector_type(8)))  __bf16 v8bf;
typedef __attribute__((ext_vector_type(8)))  float  v8f;
typedef __attribute__((ext_vector_type(16))) float  v16f;

__device__ __forceinline__ float sigmoidf_(float x) {
    return 1.0f / (1.0f + expf(-x));
}

// ---------------- fp32 -> bf16 convert (grid-strided) ----------------
__global__ void k_cvt_bf16(const float* __restrict__ src, bf16_t* __restrict__ dst, int n) {
    int i = blockIdx.x * blockDim.x + threadIdx.x;
    int stride = gridDim.x * blockDim.x;
    for (; i < n; i += stride) dst[i] = (bf16_t)src[i];
}

// ---------------- zero init of hidden state ----------------
__global__ void k_zero_h(float* __restrict__ hf, bf16_t* __restrict__ hb) {
    int i = blockIdx.x * blockDim.x + threadIdx.x;
    if (i < HIDDEN) { hf[i] = 0.0f; hb[i] = (bf16_t)0.0f; }
}

// ---------------- big GEMM: GIT[t][m] = sum_k Wih[m,k] * X[t,k] + bih[m] ----------------
// one wave per 16x16 output tile; K=4096 in chunks of 32; fp32 in, bf16 WMMA.
__global__ void k_gemm_git(const float* __restrict__ Wih, const float* __restrict__ X,
                           const float* __restrict__ bih, float* __restrict__ GIT) {
    const int K = INPUT_SIZE;
    int wave = (int)((blockIdx.x * blockDim.x + threadIdx.x) >> 5);
    int lane = threadIdx.x & 31;
    int tn = wave & 31;          // N tiles: SEQ_LEN/16 = 32
    int tm = wave >> 5;          // M tiles: GATES/16  = 384
    int grp = lane >> 4;         // 0 or 1 (lane half)
    int lm  = lane & 15;
    int row  = tm * 16 + lm;     // A row handled by this lane
    int ncol = tn * 16 + lm;     // B/D column handled by this lane
    const float* ap = Wih + (size_t)row  * K + grp * 8;
    const float* bp = X   + (size_t)ncol * K + grp * 8;
    v8f acc = {};
    for (int kc = 0; kc < K; kc += 32) {
        v8f  alo = *(const v8f*) (ap + kc);        // K = ks .. ks+7
        v8f  ahi = *(const v8f*) (ap + kc + 16);   // K = ks+16 .. ks+23
        v8f  blo = *(const v8f*) (bp + kc);        // K = ks .. ks+7
        v8f  bhi = *(const v8f*) (bp + kc + 16);   // K = ks+16 .. ks+23
        v16bf a, b;
#pragma unroll
        for (int i = 0; i < 8; ++i) { a[i] = (bf16_t)alo[i]; a[i + 8] = (bf16_t)ahi[i]; }
#pragma unroll
        for (int i = 0; i < 8; ++i) { b[i] = (bf16_t)blo[i]; b[i + 8] = (bf16_t)bhi[i]; }
        acc = __builtin_amdgcn_wmma_f32_16x16x32_bf16(false, a, false, b,
                                                      (short)0, acc, false, false);
    }
    // D layout: lane half 0 -> rows tm*16+0..7, half 1 -> rows tm*16+8..15; N = lane&15
#pragma unroll
    for (int r = 0; r < 8; ++r) {
        int ro = tm * 16 + grp * 8 + r;
        GIT[(size_t)ncol * GATES + ro] = acc[r] + bih[ro];
    }
}

// ---------------- bf16 GEMV: y[M] = W[MxK] @ x[K] (no bias) ----------------
// one wave per 16-row strip; x broadcast into all 16 B columns.
__global__ void k_gemv_bf16(const bf16_t* __restrict__ W, const bf16_t* __restrict__ x,
                            float* __restrict__ y, int K) {
    int wave = (int)((blockIdx.x * blockDim.x + threadIdx.x) >> 5);
    int lane = threadIdx.x & 31;
    int grp = lane >> 4;
    int lm  = lane & 15;
    int tmb = wave * 16;
    const bf16_t* wp = W + (size_t)(tmb + lm) * K + grp * 8;
    const bf16_t* xp = x + grp * 8;
    v8f acc = {};
    for (int kc = 0; kc < K; kc += 32) {
        v8bf lo = *(const v8bf*)(wp + kc);
        v8bf hi = *(const v8bf*)(wp + kc + 16);
        v16bf a = __builtin_shufflevector(lo, hi, 0, 1, 2, 3, 4, 5, 6, 7,
                                                  8, 9, 10, 11, 12, 13, 14, 15);
        v8bf xlo = *(const v8bf*)(xp + kc);
        v8bf xhi = *(const v8bf*)(xp + kc + 16);
        v16bf b = __builtin_shufflevector(xlo, xhi, 0, 1, 2, 3, 4, 5, 6, 7,
                                                    8, 9, 10, 11, 12, 13, 14, 15);   // same for all lanes in half -> broadcast
        acc = __builtin_amdgcn_wmma_f32_16x16x32_bf16(false, a, false, b,
                                                      (short)0, acc, false, false);
    }
    if (lm == 0) {                       // every D column equals y; store column 0 once
        int rbase = tmb + grp * 8;
#pragma unroll
        for (int r = 0; r < 8; ++r) y[rbase + r] = acc[r];
    }
}

// ---------------- encoder GRU gate update ----------------
__global__ void k_gate_enc(const float* __restrict__ GIT, int t,
                           const float* __restrict__ gh, const float* __restrict__ bhh,
                           float* __restrict__ hf, bf16_t* __restrict__ hb) {
    int i = blockIdx.x * blockDim.x + threadIdx.x;
    if (i >= HIDDEN) return;
    const float* gi = GIT + (size_t)t * GATES;   // bih already folded in
    float ir  = gi[i];
    float iz  = gi[i + HIDDEN];
    float inn = gi[i + 2 * HIDDEN];
    float hr  = gh[i]              + bhh[i];
    float hz  = gh[i + HIDDEN]     + bhh[i + HIDDEN];
    float hn  = gh[i + 2 * HIDDEN] + bhh[i + 2 * HIDDEN];
    float r = sigmoidf_(ir + hr);
    float z = sigmoidf_(iz + hz);
    float n = tanhf(inn + r * hn);
    float hnew = (1.0f - z) * n + z * hf[i];
    hf[i] = hnew;
    hb[i] = (bf16_t)hnew;
}

// ---------------- decoder GRU gate update (x is one-hot -> column gather) ----------------
__global__ void k_gate_dec(const float* __restrict__ Wih, const float* __restrict__ bih,
                           const int* __restrict__ d_idx, int step,
                           const float* __restrict__ gh, const float* __restrict__ bhh,
                           float* __restrict__ hf, bf16_t* __restrict__ hb) {
    int i = blockIdx.x * blockDim.x + threadIdx.x;
    if (i >= HIDDEN) return;
    float ir  = bih[i];
    float iz  = bih[i + HIDDEN];
    float inn = bih[i + 2 * HIDDEN];
    if (step > 0) {                 // step 0: x = zeros -> Wih@x = 0
        int idx = d_idx[0];
        ir  += Wih[(size_t)(i)              * OUTPUT_SIZE + idx];
        iz  += Wih[(size_t)(i + HIDDEN)     * OUTPUT_SIZE + idx];
        inn += Wih[(size_t)(i + 2 * HIDDEN) * OUTPUT_SIZE + idx];
    }
    float hr  = gh[i]              + bhh[i];
    float hz  = gh[i + HIDDEN]     + bhh[i + HIDDEN];
    float hn  = gh[i + 2 * HIDDEN] + bhh[i + 2 * HIDDEN];
    float r = sigmoidf_(ir + hr);
    float z = sigmoidf_(iz + hz);
    float n = tanhf(inn + r * hn);
    float hnew = (1.0f - z) * n + z * hf[i];
    hf[i] = hnew;
    hb[i] = (bf16_t)hnew;
}

// ---------------- log_softmax + argmax over 4096 (single block) ----------------
__global__ void k_lsm(const float* __restrict__ o_pre, const float* __restrict__ b_out,
                      float* __restrict__ out, int* __restrict__ d_idx) {
    __shared__ float sv[1024];
    __shared__ int   si[1024];
    int tid = threadIdx.x;
    float vmax = -3.4e38f; int vidx = 0;
#pragma unroll
    for (int k = 0; k < 4; ++k) {
        int j = tid + k * 1024;
        float v = o_pre[j] + b_out[j];
        if (v > vmax) { vmax = v; vidx = j; }
    }
    sv[tid] = vmax; si[tid] = vidx;
    __syncthreads();
    for (int s = 512; s > 0; s >>= 1) {
        if (tid < s) {
            float v2 = sv[tid + s]; int i2 = si[tid + s];
            if (v2 > sv[tid] || (v2 == sv[tid] && i2 < si[tid])) { sv[tid] = v2; si[tid] = i2; }
        }
        __syncthreads();
    }
    float gmax = sv[0];
    int   gidx = si[0];
    __syncthreads();
    float acc = 0.0f;
#pragma unroll
    for (int k = 0; k < 4; ++k) {
        int j = tid + k * 1024;
        acc += expf(o_pre[j] + b_out[j] - gmax);
    }
    sv[tid] = acc;
    __syncthreads();
    for (int s = 512; s > 0; s >>= 1) {
        if (tid < s) sv[tid] += sv[tid + s];
        __syncthreads();
    }
    float lse = gmax + logf(sv[0]);
#pragma unroll
    for (int k = 0; k < 4; ++k) {
        int j = tid + k * 1024;
        out[j] = o_pre[j] + b_out[j] - lse;
    }
    if (tid == 0) d_idx[0] = gidx;
}

extern "C" void kernel_launch(void* const* d_in, const int* in_sizes, int n_in,
                              void* d_out, int out_size, void* d_ws, size_t ws_size,
                              hipStream_t stream) {
    (void)in_sizes; (void)n_in; (void)out_size; (void)ws_size;
    const float* input   = (const float*)d_in[0];   // [512,1,4096]
    const float* enc_Wih = (const float*)d_in[1];   // [6144,4096]
    const float* enc_Whh = (const float*)d_in[2];   // [6144,2048]
    const float* enc_bih = (const float*)d_in[3];
    const float* enc_bhh = (const float*)d_in[4];
    const float* dec_Wih = (const float*)d_in[5];   // [6144,4096]
    const float* dec_Whh = (const float*)d_in[6];   // [6144,2048]
    const float* dec_bih = (const float*)d_in[7];
    const float* dec_bhh = (const float*)d_in[8];
    const float* W_out   = (const float*)d_in[9];   // [4096,2048]
    const float* b_out   = (const float*)d_in[10];
    float* out = (float*)d_out;                     // [30,1,4096]

    char* ws = (char*)d_ws;
    size_t off = 0;
    auto alloc = [&](size_t bytes) -> void* {
        void* p = ws + off;
        off += (bytes + 255) & ~(size_t)255;
        return p;
    };
    bf16_t* encWhh_b = (bf16_t*)alloc((size_t)GATES * HIDDEN * 2);       // 25.2 MB
    bf16_t* decWhh_b = (bf16_t*)alloc((size_t)GATES * HIDDEN * 2);       // 25.2 MB
    bf16_t* Wout_b   = (bf16_t*)alloc((size_t)OUTPUT_SIZE * HIDDEN * 2); // 16.8 MB
    float*  GIT      = (float*) alloc((size_t)SEQ_LEN * GATES * 4);      // 12.6 MB
    float*  gh       = (float*) alloc((size_t)GATES * 4);
    float*  hf       = (float*) alloc((size_t)HIDDEN * 4);
    bf16_t* hb       = (bf16_t*)alloc((size_t)HIDDEN * 2);
    float*  o_pre    = (float*) alloc((size_t)OUTPUT_SIZE * 4);
    int*    d_idx    = (int*)   alloc(16);

    // 1) weight precision conversion (hot weights -> bf16, resident in 192MB L2)
    k_cvt_bf16<<<4096, 256, 0, stream>>>(enc_Whh, encWhh_b, GATES * HIDDEN);
    k_cvt_bf16<<<4096, 256, 0, stream>>>(dec_Whh, decWhh_b, GATES * HIDDEN);
    k_cvt_bf16<<<4096, 256, 0, stream>>>(W_out,   Wout_b,   OUTPUT_SIZE * HIDDEN);
    k_zero_h<<<HIDDEN / 256, 256, 0, stream>>>(hf, hb);

    // 2) all encoder input-projections as one big WMMA GEMM:
    //    GIT[512][6144] = (enc_Wih @ X^T)^T + bih
    {
        int waves = (GATES / 16) * (SEQ_LEN / 16);   // 384*32 = 12288
        k_gemm_git<<<waves * 32 / 256, 256, 0, stream>>>(enc_Wih, input, enc_bih, GIT);
    }

    // 3) sequential encoder: per step only gh = Whh @ h (bf16 WMMA GEMV) + gates
    for (int t = 0; t < SEQ_LEN; ++t) {
        k_gemv_bf16<<<GATES / 128, 256, 0, stream>>>(encWhh_b, hb, gh, HIDDEN);
        k_gate_enc<<<HIDDEN / 256, 256, 0, stream>>>(GIT, t, gh, enc_bhh, hf, hb);
    }

    // 4) decoder: gh GEMV, gate update (one-hot x -> column gather), output GEMV, log_softmax+argmax
    for (int s = 0; s < STEPS; ++s) {
        k_gemv_bf16<<<GATES / 128, 256, 0, stream>>>(decWhh_b, hb, gh, HIDDEN);
        k_gate_dec<<<HIDDEN / 256, 256, 0, stream>>>(dec_Wih, dec_bih, d_idx, s,
                                                     gh, dec_bhh, hf, hb);
        k_gemv_bf16<<<OUTPUT_SIZE / 128, 256, 0, stream>>>(Wout_b, hb, o_pre, HIDDEN);
        k_lsm<<<1, 1024, 0, stream>>>(o_pre, b_out, out + (size_t)s * OUTPUT_SIZE, d_idx);
    }
}
